// LSTM_22256520528506
// MI455X (gfx1250) — matrix operand order repaired and checked by reading, not compiled, not run on hardware
//
#include <hip/hip_runtime.h>

// LSTM: B=64, T=1024, D=512, H=512.
// Persistent kernel: 32 blocks x 128 threads (4 waves). Block b owns hidden
// units j0=16b..j0+15 across all 4 gates; wave m owns batch rows 16m..16m+15.
// W slice (64 cols x 1024 K, bf16) resident in LDS for the entire T loop.
// c-state lives in registers for the entire run. Grid barrier between steps.
// x is pre-packed to bf16 in a parallel prologue (if workspace allows), so the
// sequential step loop does only b128 loads + WMMA + gate math.

#define BN 64
#define TT 1024
#define DD 512
#define HH 512
#define NBLK 32
#define LDSROW 1032          // 1024 bf16 + 8 pad elements per LDS row
#define LDSBYTES (64 * LDSROW * 2)

// workspace layout
#define OFF_BAR   (4u << 20)                 // after 4MB Wt
#define OFF_HB    (OFF_BAR + 256u)           // two bf16 h buffers (2*64KB)
#define OFF_XB    (8u << 20)                 // bf16 x copy (64MB)
#define NEED_BIG  ((size_t)(8u << 20) + (size_t)BN * TT * DD * 2u)

typedef __attribute__((ext_vector_type(16))) __bf16        v16bf;
typedef __attribute__((ext_vector_type(8)))  float         v8f;
typedef __attribute__((ext_vector_type(8)))  unsigned int  v8u;

__device__ __forceinline__ unsigned bf16u(float f) {
  unsigned u = __builtin_bit_cast(unsigned, f);
  return (u + 0x7FFFu + ((u >> 16) & 1u)) >> 16;   // RNE fp32->bf16
}
__device__ __forceinline__ unsigned pkbf(float a, float b) {
  return bf16u(a) | (bf16u(b) << 16);
}
__device__ __forceinline__ float fsig(float x) {
  return __builtin_amdgcn_rcpf(1.0f + __expf(-x));
}
__device__ __forceinline__ float ftanh(float x) {
  return 1.0f - 2.0f * __builtin_amdgcn_rcpf(__expf(2.0f * x) + 1.0f);
}

// ---------------- prologue kernels ----------------

// Pack W = concat([WI,WF,WO,WG], axis=1) transposed to Wt[n][k] bf16,
// n = gate*512 + j (0..2047), k = 0..1023. Coalesced bf16 writes.
__global__ void lstm_pack_w(const float* __restrict__ WI, const float* __restrict__ WF,
                            const float* __restrict__ WO, const float* __restrict__ WG,
                            unsigned short* __restrict__ Wt) {
  int idx = blockIdx.x * blockDim.x + threadIdx.x;   // n*1024 + k
  if (idx >= 2048 * 1024) return;
  int n = idx >> 10;
  int k = idx & 1023;
  int gate = n >> 9;
  int j = n & 511;
  const float* Wg = (gate == 0) ? WI : (gate == 1) ? WF : (gate == 2) ? WO : WG;
  Wt[idx] = (unsigned short)bf16u(Wg[k * 512 + j]);
}

// Pack x (B,T,D) fp32 -> bf16, same layout. Fully parallel, bandwidth-trivial.
__global__ void lstm_pack_x(const float* __restrict__ x, unsigned* __restrict__ xb) {
  size_t p = (size_t)blockIdx.x * blockDim.x + threadIdx.x;   // pair index
  if (p >= (size_t)BN * TT * DD / 2) return;
  float2 f = ((const float2*)x)[p];
  xb[p] = pkbf(f.x, f.y);
}

__global__ void lstm_zero(unsigned* __restrict__ p, int n) {
  int i = blockIdx.x * blockDim.x + threadIdx.x;
  if (i < n) p[i] = 0u;
}

// ---------------- grid barrier ----------------

__device__ __forceinline__ void grid_barrier(unsigned* bar) {
  __syncthreads();
  if (threadIdx.x == 0) {
    __threadfence();
    unsigned gen  = __hip_atomic_load(&bar[1], __ATOMIC_RELAXED, __HIP_MEMORY_SCOPE_AGENT);
    unsigned prev = __hip_atomic_fetch_add(&bar[0], 1u, __ATOMIC_ACQ_REL, __HIP_MEMORY_SCOPE_AGENT);
    if (prev == (unsigned)gridDim.x - 1u) {
      __hip_atomic_store(&bar[0], 0u, __ATOMIC_RELAXED, __HIP_MEMORY_SCOPE_AGENT);
      __hip_atomic_store(&bar[1], gen + 1u, __ATOMIC_RELEASE, __HIP_MEMORY_SCOPE_AGENT);
    } else {
      while (__hip_atomic_load(&bar[1], __ATOMIC_ACQUIRE, __HIP_MEMORY_SCOPE_AGENT) == gen) {
        __builtin_amdgcn_s_sleep(2);
      }
    }
  }
  __syncthreads();
}

// ---------------- main persistent kernel ----------------

template <bool PREPACKED>
__global__ void __launch_bounds__(128)
lstm_persistent(const float* __restrict__ x,
                const unsigned short* __restrict__ xb,
                const float* __restrict__ bI, const float* __restrict__ bF,
                const float* __restrict__ bO, const float* __restrict__ bG,
                const unsigned short* __restrict__ Wt,
                unsigned short* __restrict__ hb0,
                unsigned short* __restrict__ hb1,
                unsigned* __restrict__ bar,
                float* __restrict__ out) {
  extern __shared__ unsigned short lds[];
  const int tid    = threadIdx.x;
  const int lane   = tid & 31;
  const int wave   = tid >> 5;        // M-tile id 0..3
  const int lanelo = lane & 15;
  const int hih    = lane >> 4;       // lane half
  const int j0     = blockIdx.x * 16; // hidden-unit base for this block

  // Stage W slice into LDS: local row nl = gate*16 + jl  ->  global n = gate*512 + j0 + jl.
  for (int i = tid; i < 64 * 128; i += 128) {       // 8192 uint4 copies
    int nl = i >> 7;
    int c  = i & 127;
    int ng = (nl >> 4) * 512 + j0 + (nl & 15);
    uint4 v = ((const uint4*)(Wt + (size_t)ng * 1024))[c];
    *((uint4*)(lds + nl * LDSROW) + c) = v;
  }
  __syncthreads();

  float biasv[4];
  biasv[0] = bI[j0 + lanelo]; biasv[1] = bF[j0 + lanelo];
  biasv[2] = bO[j0 + lanelo]; biasv[3] = bG[j0 + lanelo];

  float creg[8];
  #pragma unroll
  for (int r = 0; r < 8; ++r) creg[r] = 0.0f;

  const int rowA  = wave * 16 + lanelo;     // A-matrix row for this lane
  const int koffA = hih * 8;                // A: lanes16-31 start at K+8
  const int koffB = hih * 8;                // B: lanes16-31 start at K+8
  const float*          xrow  = x  + (size_t)rowA * TT * DD;
  const unsigned short* xrowb = xb + (size_t)rowA * TT * DD;
  const int jcol  = j0 + lanelo;            // C/D column for this lane

  for (int t = 0; t < TT; ++t) {
    const unsigned short* hr = (t & 1) ? hb1 : hb0;   // h_{t-1}
    unsigned short*       hw = (t & 1) ? hb0 : hb1;   // h_t

    v8f acc[4];
    #pragma unroll
    for (int g = 0; g < 4; ++g)
      #pragma unroll
      for (int r = 0; r < 8; ++r) acc[g][r] = biasv[g];

    // ---- x part of gate_in: K = 0..511 ----
    #pragma unroll 4
    for (int kb = 0; kb < 16; ++kb) {
      const int k0 = kb * 32 + koffA;
      v16bf af;
      if (PREPACKED) {
        const unsigned short* xp = xrowb + (size_t)t * DD + k0;
        uint4 a0 = *(const uint4*)(xp);
        uint4 a1 = *(const uint4*)(xp + 16);
        v8u au;
        au[0] = a0.x; au[1] = a0.y; au[2] = a0.z; au[3] = a0.w;
        au[4] = a1.x; au[5] = a1.y; au[6] = a1.z; au[7] = a1.w;
        af = __builtin_bit_cast(v16bf, au);
      } else {
        const float* xt = xrow + (size_t)t * DD;
        float4 f0 = *(const float4*)(xt + k0);
        float4 f1 = *(const float4*)(xt + k0 + 4);
        float4 f2 = *(const float4*)(xt + k0 + 16);
        float4 f3 = *(const float4*)(xt + k0 + 20);
        v8u au;
        au[0] = pkbf(f0.x, f0.y); au[1] = pkbf(f0.z, f0.w);
        au[2] = pkbf(f1.x, f1.y); au[3] = pkbf(f1.z, f1.w);
        au[4] = pkbf(f2.x, f2.y); au[5] = pkbf(f2.z, f2.w);
        au[6] = pkbf(f3.x, f3.y); au[7] = pkbf(f3.z, f3.w);
        af = __builtin_bit_cast(v16bf, au);
      }
      const int kB = kb * 32 + koffB;
      #pragma unroll
      for (int g = 0; g < 4; ++g) {
        const uint4* bp = (const uint4*)(lds + (g * 16 + lanelo) * LDSROW + kB);
        uint4 b0 = bp[0], b1 = bp[2];
        v8u bu;
        bu[0] = b0.x; bu[1] = b0.y; bu[2] = b0.z; bu[3] = b0.w;
        bu[4] = b1.x; bu[5] = b1.y; bu[6] = b1.z; bu[7] = b1.w;
        v16bf bfv = __builtin_bit_cast(v16bf, bu);
        acc[g] = __builtin_amdgcn_wmma_f32_16x16x32_bf16(
            false, af, false, bfv, (short)0, acc[g], false, false);
      }
    }

    // ---- h part of gate_in: K = 512..1023 (bf16, direct loads) ----
    #pragma unroll 4
    for (int kb = 0; kb < 16; ++kb) {
      const int kh = kb * 32 + koffA;
      const unsigned short* hp = hr + (size_t)rowA * HH + kh;
      uint4 h0 = *(const uint4*)(hp);
      uint4 h1 = *(const uint4*)(hp + 16);
      v8u au;
      au[0] = h0.x; au[1] = h0.y; au[2] = h0.z; au[3] = h0.w;
      au[4] = h1.x; au[5] = h1.y; au[6] = h1.z; au[7] = h1.w;
      v16bf af = __builtin_bit_cast(v16bf, au);
      const int kB = 512 + kb * 32 + koffB;
      #pragma unroll
      for (int g = 0; g < 4; ++g) {
        const uint4* bp = (const uint4*)(lds + (g * 16 + lanelo) * LDSROW + kB);
        uint4 b0 = bp[0], b1 = bp[2];
        v8u bu;
        bu[0] = b0.x; bu[1] = b0.y; bu[2] = b0.z; bu[3] = b0.w;
        bu[4] = b1.x; bu[5] = b1.y; bu[6] = b1.z; bu[7] = b1.w;
        v16bf bfv = __builtin_bit_cast(v16bf, bu);
        acc[g] = __builtin_amdgcn_wmma_f32_16x16x32_bf16(
            false, af, false, bfv, (short)0, acc[g], false, false);
      }
    }

    // ---- gates -> c, h (C/D layout: row = 16*wave + r + 8*hih, col = jcol) ----
    #pragma unroll
    for (int r = 0; r < 8; ++r) {
      float iv = fsig(acc[0][r]);
      float fv = fsig(acc[1][r]);
      float ov = fsig(acc[2][r]);
      float gv = ftanh(acc[3][r]);
      float c  = fv * creg[r] + iv * gv;
      creg[r]  = c;
      float hv = ov * ftanh(c);
      const int b = wave * 16 + r + hih * 8;
      out[((size_t)b * TT + t) * HH + jcol] = hv;
      hw[(size_t)b * HH + jcol] = (unsigned short)bf16u(hv);
      if (t == TT - 1)
        out[(size_t)BN * TT * HH + (size_t)b * HH + jcol] = hv;
    }

    grid_barrier(bar);
  }

  // final c
  #pragma unroll
  for (int r = 0; r < 8; ++r) {
    const int b = wave * 16 + r + hih * 8;
    out[(size_t)BN * TT * HH + (size_t)BN * HH + (size_t)b * HH + jcol] = creg[r];
  }
}

// ---------------- host ----------------

extern "C" void kernel_launch(void* const* d_in, const int* in_sizes, int n_in,
                              void* d_out, int out_size, void* d_ws, size_t ws_size,
                              hipStream_t stream) {
  (void)in_sizes; (void)n_in; (void)out_size;
  const float* x  = (const float*)d_in[0];
  const float* WI = (const float*)d_in[1];
  const float* bI = (const float*)d_in[2];
  const float* WF = (const float*)d_in[3];
  const float* bF = (const float*)d_in[4];
  const float* WO = (const float*)d_in[5];
  const float* bO = (const float*)d_in[6];
  const float* WG = (const float*)d_in[7];
  const float* bG = (const float*)d_in[8];
  float* out = (float*)d_out;

  char* ws = (char*)d_ws;
  unsigned short* Wt  = (unsigned short*)ws;                 // [0, 4MB)
  unsigned*       bar = (unsigned*)(ws + OFF_BAR);
  unsigned short* hb0 = (unsigned short*)(ws + OFF_HB);
  unsigned short* hb1 = hb0 + BN * HH;
  unsigned short* xb  = (unsigned short*)(ws + OFF_XB);      // 64MB bf16 x

  const bool big = (ws_size >= NEED_BIG);                    // constant per run

  (void)hipFuncSetAttribute((const void*)lstm_persistent<true>,
                            hipFuncAttributeMaxDynamicSharedMemorySize, LDSBYTES);
  (void)hipFuncSetAttribute((const void*)lstm_persistent<false>,
                            hipFuncAttributeMaxDynamicSharedMemorySize, LDSBYTES);

  lstm_pack_w<<<(2048 * 1024) / 256, 256, 0, stream>>>(WI, WF, WO, WG, Wt);

  int nzero = (256 + 2 * BN * HH * 2) / 4;   // barrier words + both h buffers
  lstm_zero<<<(nzero + 255) / 256, 256, 0, stream>>>(bar, nzero);

  if (big) {
    const size_t npairs = (size_t)BN * TT * DD / 2;
    lstm_pack_x<<<(unsigned)((npairs + 255) / 256), 256, 0, stream>>>(x, (unsigned*)xb);
    lstm_persistent<true><<<NBLK, 128, LDSBYTES, stream>>>(
        x, xb, bI, bF, bO, bG, Wt, hb0, hb1, bar, out);
  } else {
    lstm_persistent<false><<<NBLK, 128, LDSBYTES, stream>>>(
        x, xb, bI, bF, bO, bG, Wt, hb0, hb1, bar, out);
  }
}
